// StyleConv2d_5351529251373
// MI455X (gfx1250) — matrix operand order repaired and checked by reading, not compiled, not run on hardware
//
#include <hip/hip_runtime.h>
#include <hip/hip_bf16.h>

// ---------------- problem constants ----------------
#define B_      16
#define CIN_    512
#define COUT_   512
#define H_      64
#define W_      64
#define HW_     4096          // 64*64
#define SDIM_   32
#define EPS_    1e-8f

// ---------------- WMMA vector types ----------------
typedef __attribute__((ext_vector_type(16))) __bf16 v16bf;
typedef __attribute__((ext_vector_type(8)))  float  v8f;

union U16BF {
    uint4 q[2];
    v16bf v;
};

__device__ __forceinline__ v16bf make_frag(uint4 lo, uint4 hi) {
    U16BF u; u.q[0] = lo; u.q[1] = hi; return u.v;
}

// round-to-nearest-even f32 -> bf16 (bit pattern as ushort) — weights only
__device__ __forceinline__ unsigned short f2bf(float f) {
    union { float f; unsigned int u; } v; v.f = f;
    unsigned int u = v.u;
    u += 0x7FFFu + ((u >> 16) & 1u);
    return (unsigned short)(u >> 16);
}

// pack two f32 -> {lo16 = bf16(lo), hi16 = bf16(hi)} in one dword.
// single v_perm_b32: dst = {hi[31:16], lo[31:16]} (bit truncation)
__device__ __forceinline__ unsigned int pkbf(float lo, float hi) {
    unsigned int ulo = __builtin_bit_cast(unsigned int, lo);
    unsigned int uhi = __builtin_bit_cast(unsigned int, hi);
#if __has_builtin(__builtin_amdgcn_perm)
    return __builtin_amdgcn_perm(uhi, ulo, 0x07060302u);
#else
    return (uhi & 0xFFFF0000u) | (ulo >> 16);
#endif
}

// ---------------------------------------------------------------
// Kernel 1: s[b, cin] = style[b,:] . mod_w[cin,:] + mod_b[cin]
// grid: B_ blocks, 512 threads
// ---------------------------------------------------------------
__global__ void mod_kernel(const float* __restrict__ style,
                           const float* __restrict__ mod_w,
                           const float* __restrict__ mod_b,
                           float* __restrict__ s_out) {
    __shared__ float st[SDIM_];
    const int b = blockIdx.x;
    const int t = threadIdx.x;           // cin
    if (t < SDIM_) st[t] = style[b * SDIM_ + t];
    __syncthreads();
    float acc = 0.0f;
    const float* mw = mod_w + t * SDIM_;
    #pragma unroll
    for (int d = 0; d < SDIM_; ++d) acc = fmaf(st[d], mw[d], acc);
    s_out[b * CIN_ + t] = acc + mod_b[t];
}

// ---------------------------------------------------------------
// Kernel 2: demodulated bf16 weights
//   w[b,cout,cin] = conv_w[cout,cin]*s[b,cin];  demod = rsqrt(sum w^2 + eps)
// grid: (COUT_, B_) blocks, 256 threads; thread covers cin = t, t+256
// ---------------------------------------------------------------
__global__ void wmod_kernel(const float* __restrict__ conv_w,
                            const float* __restrict__ s_in,
                            unsigned short* __restrict__ wbf) {
    __shared__ float red[256];
    const int cout = blockIdx.x;
    const int b    = blockIdx.y;
    const int t    = threadIdx.x;

    const float* cw = conv_w + (size_t)cout * CIN_;
    const float* sv = s_in   + (size_t)b * CIN_;

    float v0 = cw[t]       * sv[t];
    float v1 = cw[t + 256] * sv[t + 256];

    red[t] = v0 * v0 + v1 * v1;
    __syncthreads();
    for (int sft = 128; sft > 0; sft >>= 1) {
        if (t < sft) red[t] += red[t + sft];
        __syncthreads();
    }
    const float demod = rsqrtf(red[0] + EPS_);

    unsigned short* wr = wbf + ((size_t)(b * COUT_ + cout)) * CIN_;
    wr[t]       = f2bf(v0 * demod);
    wr[t + 256] = f2bf(v1 * demod);
}

// ---------------------------------------------------------------
// Kernel 3: per-sample GEMM  Y[b, cout, p] = sum_cin Wb[cout,cin] * X[b,cin,p]
// bf16 WMMA, f32 accumulate, double-buffered LDS X tile.
// grid: (HW_/64 = 64, COUT_/128 = 4, B_) ; block 256 threads = 8 waves
// wave w -> cout rows [cout0 + 16w, +16), pixel subtiles {0..3} of 16
// ---------------------------------------------------------------
#define KTILE 32
#define PXT   64
#define XS    40              // LDS row stride in bf16 (80 B: 16B aligned, bank-safe)

__global__ void gemm_kernel(const float* __restrict__ x,
                            const unsigned short* __restrict__ wbf,
                            float* __restrict__ y) {
    __shared__ unsigned short xt[2][PXT * XS];   // transposed X tiles: [pixel][cin]

    const int t     = threadIdx.x;
    const int wave  = t >> 5;
    const int lane  = t & 31;
    const int m     = lane & 15;
    const int half  = lane >> 4;

    const int px0   = blockIdx.x * PXT;
    const int cout0 = blockIdx.y * 128;
    const int b     = blockIdx.z;

    // staging: thread covers 2 consecutive pixels x 4 consecutive cin
    const int n0 = (t & 31) * 2;               // pixel pair base 0..62
    const int kq = (t >> 5) * 4;               // cin quad base 0..28

    const float* xbase = x + ((size_t)b * CIN_ + kq) * HW_ + px0 + n0;
    const unsigned short* wrow =
        wbf + ((size_t)(b * COUT_ + cout0 + wave * 16 + m)) * CIN_;

    v8f acc0 = {}, acc1 = {}, acc2 = {}, acc3 = {};

    // ---- prologue: stage k0 = 0 into buffer 0 ----
    float2 r0 = *(const float2*)(xbase);
    float2 r1 = *(const float2*)(xbase + HW_);
    float2 r2 = *(const float2*)(xbase + 2 * HW_);
    float2 r3 = *(const float2*)(xbase + 3 * HW_);
    {
        unsigned int* d0 = (unsigned int*)&xt[0][n0 * XS + kq];
        unsigned int* d1 = (unsigned int*)&xt[0][(n0 + 1) * XS + kq];
        d0[0] = pkbf(r0.x, r1.x); d0[1] = pkbf(r2.x, r3.x);
        d1[0] = pkbf(r0.y, r1.y); d1[1] = pkbf(r2.y, r3.y);
    }
    __syncthreads();

    int buf = 0;
    for (int k0 = 0; k0 < CIN_; k0 += KTILE) {
        const bool more = (k0 + KTILE) < CIN_;

        // issue next tile's global loads early (overlap with WMMAs)
        if (more) {
            const float* xp = xbase + (size_t)(k0 + KTILE) * HW_;
            r0 = *(const float2*)(xp);
            r1 = *(const float2*)(xp + HW_);
            r2 = *(const float2*)(xp + 2 * HW_);
            r3 = *(const float2*)(xp + 3 * HW_);
        }

        // ---- A fragment: direct global b128 pair (16-bit A 16x32 layout) ----
        const uint4* wp = (const uint4*)(wrow + k0);
        const v16bf a = make_frag(wp[2 * half], wp[2 * half + 1]);

        // ---- B fragments from LDS (32x16 B layout): row = pixel, contig cin ----
        const unsigned short* xb = xt[buf];
        const uint4* bp0 = (const uint4*)(&xb[(m)      * XS + half * 16]);
        const uint4* bp1 = (const uint4*)(&xb[(16 + m) * XS + half * 16]);
        const uint4* bp2 = (const uint4*)(&xb[(32 + m) * XS + half * 16]);
        const uint4* bp3 = (const uint4*)(&xb[(48 + m) * XS + half * 16]);

        acc0 = __builtin_amdgcn_wmma_f32_16x16x32_bf16(
                   false, a, false, make_frag(bp0[0], bp0[1]), (short)0, acc0, false, false);
        acc1 = __builtin_amdgcn_wmma_f32_16x16x32_bf16(
                   false, a, false, make_frag(bp1[0], bp1[1]), (short)0, acc1, false, false);
        acc2 = __builtin_amdgcn_wmma_f32_16x16x32_bf16(
                   false, a, false, make_frag(bp2[0], bp2[1]), (short)0, acc2, false, false);
        acc3 = __builtin_amdgcn_wmma_f32_16x16x32_bf16(
                   false, a, false, make_frag(bp3[0], bp3[1]), (short)0, acc3, false, false);

        // ---- stage next tile into the other buffer ----
        if (more) {
            unsigned int* d0 = (unsigned int*)&xt[buf ^ 1][n0 * XS + kq];
            unsigned int* d1 = (unsigned int*)&xt[buf ^ 1][(n0 + 1) * XS + kq];
            d0[0] = pkbf(r0.x, r1.x); d0[1] = pkbf(r2.x, r3.x);
            d1[0] = pkbf(r0.y, r1.y); d1[1] = pkbf(r2.y, r3.y);
        }
        __syncthreads();
        buf ^= 1;
    }

    // ---- epilogue: C/D layout -> Y[b, cout, pixel] ----
    // VGPR r holds M = r + 8*half, N = lane%16
    #pragma unroll
    for (int r = 0; r < 8; ++r) {
        const int mm = r + 8 * half;
        float* yr = y + (size_t)(b * COUT_ + cout0 + wave * 16 + mm) * HW_ + px0 + m;
        yr[0]  = acc0[r];
        yr[16] = acc1[r];
        yr[32] = acc2[r];
        yr[48] = acc3[r];
    }
}

// ---------------------------------------------------------------
// Kernel 4: bilinear x2 upsample, half-pixel centers, edge clamp.
// Each thread emits a 2x2 output quad from a 3x3 input neighborhood.
// weights: even out = {0.25, 0.75}, odd out = {0.75, 0.25} per axis
// ---------------------------------------------------------------
__global__ void upsample_kernel(const float* __restrict__ y,
                                float* __restrict__ out) {
    const size_t idx = (size_t)blockIdx.x * 256 + threadIdx.x;
    const int j  = (int)(idx & 63);            // input col
    const int i  = (int)((idx >> 6) & 63);     // input row
    const size_t bc = idx >> 12;               // b*COUT + c

    const float* src = y + bc * (size_t)HW_;
    const int im = max(i - 1, 0), ip = min(i + 1, H_ - 1);
    const int jm = max(j - 1, 0), jp = min(j + 1, W_ - 1);

    const float vmm = src[im * W_ + jm], vm0 = src[im * W_ + j], vmp = src[im * W_ + jp];
    const float v0m = src[i  * W_ + jm], v00 = src[i  * W_ + j], v0p = src[i  * W_ + jp];
    const float vpm = src[ip * W_ + jm], vp0 = src[ip * W_ + j], vpp = src[ip * W_ + jp];

    const float wa = 0.5625f, wb = 0.1875f, wc = 0.0625f;
    float2 top, bot;
    top.x = wa * v00 + wb * (vm0 + v0m) + wc * vmm;   // out(2i,   2j)
    top.y = wa * v00 + wb * (vm0 + v0p) + wc * vmp;   // out(2i,   2j+1)
    bot.x = wa * v00 + wb * (vp0 + v0m) + wc * vpm;   // out(2i+1, 2j)
    bot.y = wa * v00 + wb * (vp0 + v0p) + wc * vpp;   // out(2i+1, 2j+1)

    float* dst = out + bc * (size_t)(128 * 128) + (size_t)(2 * i) * 128 + 2 * j;
    *(float2*)dst         = top;
    *(float2*)(dst + 128) = bot;
}

// ---------------------------------------------------------------
extern "C" void kernel_launch(void* const* d_in, const int* in_sizes, int n_in,
                              void* d_out, int out_size, void* d_ws, size_t ws_size,
                              hipStream_t stream) {
    (void)in_sizes; (void)n_in; (void)out_size; (void)ws_size;

    const float* x      = (const float*)d_in[0];   // [16,512,64,64]
    const float* style  = (const float*)d_in[1];   // [16,32]
    const float* conv_w = (const float*)d_in[2];   // [512,512,1,1]
    const float* mod_w  = (const float*)d_in[3];   // [512,32]
    const float* mod_b  = (const float*)d_in[4];   // [512]
    float* out = (float*)d_out;                    // [16,512,128,128]

    // workspace layout
    char* ws = (char*)d_ws;
    float*          s_mod = (float*)ws;                                     // 32 KB
    unsigned short* wbf   = (unsigned short*)(ws + 32768);                  // 8 MB
    float*          ybuf  = (float*)(ws + 32768 + (size_t)B_*COUT_*CIN_*2); // 134 MB

    mod_kernel <<<B_, 512, 0, stream>>>(style, mod_w, mod_b, s_mod);
    wmod_kernel<<<dim3(COUT_, B_), 256, 0, stream>>>(conv_w, s_mod, wbf);
    gemm_kernel<<<dim3(HW_ / PXT, COUT_ / 128, B_), 256, 0, stream>>>(x, wbf, ybuf);

    const size_t quads = (size_t)B_ * COUT_ * H_ * W_;    // one thread per 2x2 out quad
    upsample_kernel<<<(unsigned)(quads / 256), 256, 0, stream>>>(ybuf, out);
}
